// QuantConv2D_64304250355995
// MI455X (gfx1250) — 8-bit matrix operand order repaired and checked by reading, compiled, not run on hardware
//
#include <hip/hip_runtime.h>


typedef __attribute__((ext_vector_type(8))) int   v8i;
typedef __attribute__((ext_vector_type(4))) int   v4i;
typedef __attribute__((ext_vector_type(4))) float v4f;

#define BATCH 32
#define CH    64
#define HH    112
#define WW    112
#define OCH   64

#if __has_builtin(__builtin_amdgcn_global_load_async_to_lds_b128) && \
    __has_builtin(__builtin_amdgcn_s_wait_asynccnt)
#define HAVE_ASYNC_LDS 1
#else
#define HAVE_ASYNC_LDS 0
#endif

__global__ void quant_x_kernel(const float* __restrict__ x,
                               unsigned char* __restrict__ xq) {
  int idx = blockIdx.x * 256 + threadIdx.x;
  int w = idx % WW;
  int h = (idx / WW) % HH;
  int c = (idx / (WW * HH)) % CH;
  int b =  idx / (WW * HH * CH);
  float v = __builtin_nontemporal_load(&x[idx]);
  v = rintf(v);
  v = fminf(fmaxf(v, 0.0f), 255.0f);
  int chunk = c >> 3, r = c & 7;
  int j = (chunk & 1) ? (32 + (chunk >> 1) * 8 + r)
                      : (     (chunk >> 1) * 8 + r);
  xq[(((size_t)(b * HH + h)) * WW + w) * 64 + j] = (unsigned char)(int)v;
}

__global__ void quant_w_kernel(const float* __restrict__ wt,
                               signed char* __restrict__ wq) {
  int idx = blockIdx.x * 256 + threadIdx.x;
  int j  = idx & 63;
  int oc = (idx >> 6) & 63;
  int p  = idx >> 12;
  int k = ((((j & 31) >> 3) * 2 + (j >> 5)) << 3) | (j & 7);
  int kh = p / 3, kw = p % 3;
  float v = rintf(wt[((oc * 64 + k) * 3 + kh) * 3 + kw]);
  v = fminf(fmaxf(v, -128.0f), 127.0f);
  wq[idx] = (signed char)(int)v;
}

__device__ __forceinline__ v8i ld_frag32(const void* p) {
  const uint4* q = (const uint4*)p;
  uint4 x0 = q[0], x1 = q[1];
  v8i r;
  r[0] = (int)x0.x; r[1] = (int)x0.y; r[2] = (int)x0.z; r[3] = (int)x0.w;
  r[4] = (int)x1.x; r[5] = (int)x1.y; r[6] = (int)x1.z; r[7] = (int)x1.w;
  return r;
}

__global__ __launch_bounds__(128, 1)
void conv_wmma_kernel(const unsigned char* __restrict__ xq,
                      const signed char*  __restrict__ wq,
                      const float*        __restrict__ bias,
                      float*              __restrict__ out) {
  __shared__ uint4 wlds[2304];
#if HAVE_ASYNC_LDS
  {
    v4i* g = (v4i*)wq;
    v4i* l = (v4i*)&wlds[0];
    for (int i = threadIdx.x; i < 2304; i += 128)
      __builtin_amdgcn_global_load_async_to_lds_b128(g + i, l + i, 0, 0);
    __builtin_amdgcn_s_wait_asynccnt(0);
  }
#else
  {
    const uint4* wg = (const uint4*)wq;
    for (int i = threadIdx.x; i < 2304; i += 128) wlds[i] = wg[i];
  }
#endif
  __syncthreads();

  const int wave = threadIdx.x >> 5;
  const int lane = threadIdx.x & 31;
  const int m    = lane & 15;
  const int half = lane >> 4;

  const int tile = blockIdx.x * 4 + wave;
  const int ow0  = (tile % 7) * 16;
  const int rest = tile / 7;
  const int oh   = rest % HH;
  const int b    = rest / HH;

  v8i acc[4] = {};
  const char* wl = (const char*)wlds;

#pragma unroll
  for (int kh = 0; kh < 3; ++kh) {
    const int ih = oh - 1 + kh;
    const bool hok = (unsigned)ih < (unsigned)HH;
    if (hok) {
      __builtin_prefetch(xq + (((size_t)(b * HH + ih)) * WW + ow0) * 64, 0, 1);
    }
#pragma unroll
    for (int kw = 0; kw < 3; ++kw) {
      const int iw = ow0 - 1 + kw + m;
      uint4 a0 = {0, 0, 0, 0}, a1 = {0, 0, 0, 0};
      if (hok && (unsigned)iw < (unsigned)WW) {
        const uint4* ap = (const uint4*)(
            xq + (((size_t)(b * HH + ih)) * WW + iw) * 64 + half * 32);
        a0 = ap[0];
        a1 = ap[1];
      }
      v8i a;
      a[0] = (int)a0.x; a[1] = (int)a0.y; a[2] = (int)a0.z; a[3] = (int)a0.w;
      a[4] = (int)a1.x; a[5] = (int)a1.y; a[6] = (int)a1.z; a[7] = (int)a1.w;

      const int p = kh * 3 + kw;
      const char* wp = wl + p * 4096 + half * 32;
#pragma unroll
      for (int nt = 0; nt < 4; ++nt) {
        v8i bf = ld_frag32(wp + (nt * 16 + m) * 64);
        acc[nt] = __builtin_amdgcn_wmma_i32_16x16x64_iu8(
            false, a, true, bf, acc[nt], false, false);
      }
    }
  }

#pragma unroll
  for (int nt = 0; nt < 4; ++nt) {
    const int oc = nt * 16 + m;
    const float bv = bias[oc];
    v4f f0, f1;
    f0.x = (float)acc[nt][0] + bv;
    f0.y = (float)acc[nt][1] + bv;
    f0.z = (float)acc[nt][2] + bv;
    f0.w = (float)acc[nt][3] + bv;
    f1.x = (float)acc[nt][4] + bv;
    f1.y = (float)acc[nt][5] + bv;
    f1.z = (float)acc[nt][6] + bv;
    f1.w = (float)acc[nt][7] + bv;
    float* op = out + (((size_t)(b * OCH + oc)) * HH + oh) * WW
                    + ow0 + half * 8;
    __builtin_nontemporal_store(f0, (v4f*)op);
    __builtin_nontemporal_store(f1, (v4f*)op + 1);
  }
}

extern "C" void kernel_launch(void* const* d_in, const int* in_sizes, int n_in,
                              void* d_out, int out_size, void* d_ws,
                              size_t ws_size, hipStream_t stream) {
  const float* x    = (const float*)d_in[0];
  const float* wt   = (const float*)d_in[1];
  const float* bias = (const float*)d_in[2];
  float* out = (float*)d_out;

  const size_t xq_bytes = (size_t)BATCH * HH * WW * 64;
  unsigned char* xq = (unsigned char*)d_ws;
  signed char*   wq = (signed char*)d_ws + xq_bytes;

  quant_x_kernel<<<100352, 256, 0, stream>>>(x, xq);
  quant_w_kernel<<<144, 256, 0, stream>>>(wt, wq);
  conv_wmma_kernel<<<6272, 128, 0, stream>>>(xq, wq, bias, out);
}
